// FCOS_9139690406412
// MI455X (gfx1250) — matrix operand order repaired and checked by reading, not compiled, not run on hardware
//
#include <hip/hip_runtime.h>

// FCOS head for MI455X (gfx1250, wave32).
// conv3x3 = implicit GEMM on v_wmma_f32_16x16x32_f16 (f16 in, f32 accum).
// - weights pre-transposed to [tap][co][ci] and staged to LDS with
//   GLOBAL_LOAD_ASYNC_TO_LDS_B128 (double buffered, issued one tap ahead)
// - activations staged once per 32-ci chunk as a halo tile; every tap reads
//   B fragments straight out of the halo with ds_load_b128
// - BN-train folded into a per-channel affine applied during f32->f16 staging

typedef __attribute__((ext_vector_type(16))) _Float16 v16h;
typedef __attribute__((ext_vector_type(8)))  _Float16 v8h;
typedef __attribute__((ext_vector_type(8)))  float    v8f;
typedef __attribute__((__vector_size__(4 * sizeof(int)))) int v4i;

#define CIN 256
#define MT  128   // co tile per workgroup (8 waves x 16)
#define NT  64    // pixel tile per workgroup
#define HALO_MAX 198  // (64/W_tile + 2) * (W_tile + 2), max over levels

// ---- gfx1250 async global->LDS (ASYNCcnt) with graceful fallback ----------
#if defined(__gfx1250__) && __has_builtin(__builtin_amdgcn_global_load_async_to_lds_b128)
#define ASYNC_COPY_B128(gsrc, ldst)                                         \
  __builtin_amdgcn_global_load_async_to_lds_b128(                           \
      (__attribute__((address_space(1))) v4i*)(gsrc),                       \
      (__attribute__((address_space(3))) v4i*)(ldst), 0, 0)
#if __has_builtin(__builtin_amdgcn_s_wait_asynccnt)
#define WAIT_ASYNC() __builtin_amdgcn_s_wait_asynccnt(0)
#else
#define WAIT_ASYNC() asm volatile("s_wait_asynccnt 0" ::: "memory")
#endif
#else
#define ASYNC_COPY_B128(gsrc, ldst) \
  (*(v8h*)(void*)(ldst) = *(const v8h*)(const void*)(gsrc))
#define WAIT_ASYNC() ((void)0)
#endif

// ---------------------------------------------------------------- utilities

// f32 [co][ci][tap] -> f16 [tap][co][ci]  (so LDS staging is contiguous b128)
__global__ __launch_bounds__(256)
void wconv_transpose_f16(const float* __restrict__ src, _Float16* __restrict__ dst) {
  int i = blockIdx.x * 256 + threadIdx.x;     // over 9*256*256
  if (i >= 9 * CIN * CIN) return;
  int ci  = i & 255;
  int co  = (i >> 8) & 255;
  int tap = i >> 16;
  dst[i] = (_Float16)src[((co << 8) + ci) * 9 + tap];
}

__global__ void init_affine_identity(float* __restrict__ aff) {
  int c = threadIdx.x;          // 256 threads
  aff[c] = 1.f;
  aff[256 + c] = 0.f;
}

// ------------------------------------------------ FPN 1x1 cross conv (+up2)

__global__ __launch_bounds__(256)
void cross_conv_kernel(const float* __restrict__ x, const float* __restrict__ w,
                       const float* __restrict__ bias, const float* __restrict__ up,
                       float* __restrict__ out, int Cin, int shift, int hasUp)
{
  const int W  = 1 << shift;
  const int HW = 1 << (2 * shift);
  int idx = blockIdx.x * 256 + threadIdx.x;
  int total = 4 * CIN * HW;
  if (idx >= total) return;
  int p  = idx & (HW - 1);
  int co = (idx >> (2 * shift)) & 255;
  int n  = idx >> (2 * shift + 8);
  const float* xp = x + (size_t)n * Cin * HW + p;
  const float* wp = w + (size_t)co * Cin;
  float acc = bias[co];
  for (int ci = 0; ci < Cin; ++ci)
    acc = fmaf(wp[ci], xp[(size_t)ci << (2 * shift)], acc);
  if (hasUp) {
    int h = p >> shift, ww = p & (W - 1);
    int upSh = shift - 1;
    acc += up[(((size_t)n * CIN + co) << (2 * upSh)) + ((h >> 1) << upSh) + (ww >> 1)];
  }
  out[idx] = acc;
}

// ------------------------------------- conv3x3 (pad 1) implicit-GEMM + WMMA
// z = relu( conv3x3( a*x + b ) + bias ), NCHW, H==W==1<<shift.
// 8 waves: wave w -> co sub-tile [co0+16w, +16), all 64 pixels (4 C tiles).

__global__ __launch_bounds__(256)
void conv3x3_bnin_relu_wmma(const float* __restrict__ x,
                            const float* __restrict__ affine,    // [512] a|b
                            const _Float16* __restrict__ w16t,   // [9][256][256]
                            const float* __restrict__ bias,      // [256]
                            float* __restrict__ z,               // [4,256,H,W]
                            int shift)
{
  const int W = 1 << shift, H = W;
  const int co0 = blockIdx.x * MT;
  const int p0  = blockIdx.y * NT;
  const int n   = blockIdx.z;

  __shared__ __align__(16) _Float16 aT[2][MT * 32];     // 2 x 8 KB weights
  __shared__ __align__(16) _Float16 bH[HALO_MAX * 32];  // 12.4 KB activations

  const int tid  = threadIdx.x;
  const int wave = tid >> 5;
  const int lane = tid & 31;
  const int m    = lane & 15;
  const int kg   = lane >> 4;

  // tile geometry: tiles are 64 consecutive flattened pixels
  const int tw    = (W < 64) ? W : 64;        // tile width
  const int twsh  = (shift < 6) ? shift : 6;
  const int Wh    = tw + 2;                   // halo width
  const int rowsH = (NT >> twsh) + 2;         // halo rows
  const int haloElems = rowsH * Wh * 32;
  const int startRow  = p0 >> shift;
  const int startCol  = p0 & (W - 1);

  v8f acc[4] = {};

  auto stageA = [&](int tap, int buf, int k0) {
    const _Float16* g = w16t + (size_t)((tap * CIN + co0) * CIN + k0)
                        + (tid >> 1) * CIN + (tid & 1) * 16;
    _Float16* l = &aT[buf][tid * 16];
    ASYNC_COPY_B128(g, l);
    ASYNC_COPY_B128(g + 8, l + 8);
  };

  for (int k0 = 0; k0 < CIN; k0 += 32) {
    __syncthreads();  // previous chunk's compute done: aT/bH reusable
    // ---- stage halo activation tile once for all 9 taps of this ci-chunk
    for (int s = tid; s < haloElems; s += 256) {
      int ci = s & 31;
      int hp = s >> 5;
      int hr = hp / Wh;
      int hc = hp - hr * Wh;
      int hh = startRow - 1 + hr;
      int ww = startCol - 1 + hc;
      float v = 0.f;
      if ((unsigned)hh < (unsigned)H && (unsigned)ww < (unsigned)W) {
        int c = k0 + ci;
        v = fmaf(affine[c], x[((n * CIN + c) << (2 * shift)) + (hh << shift) + ww],
                 affine[256 + c]);
      }
      bH[(hp << 5) + ci] = (_Float16)v;
    }
    stageA(0, 0, k0);   // async prefetch of first tap's weights
    int buf = 0;
#pragma unroll
    for (int tap = 0; tap < 9; ++tap) {
      const int dh = tap / 3 - 1;
      const int dw = tap % 3 - 1;
      WAIT_ASYNC();       // our async A(tap) landed
      __syncthreads();    // everyone's A(tap) + halo visible
      if (tap < 8) stageA(tap + 1, buf ^ 1, k0);  // overlap copy with math

      // A fragment (ISA 7.12.2 16-bit A 16x32)
      v16h afrag;
      {
        const int arow = (wave * 16 + m) << 5;
        v8h lo = *(const v8h*)&aT[buf][arow + kg * 8];
        v8h hi = *(const v8h*)&aT[buf][arow + 16 + kg * 8];
#pragma unroll
        for (int i = 0; i < 8; ++i) { afrag[i] = lo[i]; afrag[8 + i] = hi[i]; }
      }
      // B fragments straight from the halo (ci contiguous for any tap)
#pragma unroll
      for (int t = 0; t < 4; ++t) {
        const int pl = t * 16 + m;
        const int lr = pl >> twsh;
        const int lc = pl & (tw - 1);
        const int hpb = ((lr + 1 + dh) * Wh + (lc + 1 + dw)) << 5;
        v8h lo = *(const v8h*)&bH[hpb + kg * 8];
        v8h hi = *(const v8h*)&bH[hpb + 16 + kg * 8];
        v16h bf;
#pragma unroll
        for (int i = 0; i < 8; ++i) { bf[i] = lo[i]; bf[8 + i] = hi[i]; }
        acc[t] = __builtin_amdgcn_wmma_f32_16x16x32_f16(false, afrag, false, bf,
                                                        (short)0, acc[t], false, false);
      }
      buf ^= 1;
    }
  }

  // Epilogue: bias + relu, store f32.
  // C layout: VGPR r -> M=r (lanes 0-15) / M=r+8 (lanes 16-31); N = lane&15.
  const int mhi = kg * 8;
#pragma unroll
  for (int t = 0; t < 4; ++t) {
    const int p = p0 + t * 16 + m;
#pragma unroll
    for (int r = 0; r < 8; ++r) {
      const int co = co0 + wave * 16 + mhi + r;
      float v = acc[t][r] + bias[co];
      v = v > 0.f ? v : 0.f;
      z[((n * CIN + co) << (2 * shift)) + p] = v;
    }
  }
}

// -------------------------------------------- BN training stats (deterministic)

__global__ __launch_bounds__(256)
void bn_stats_kernel(const float* __restrict__ z, int HW, float* __restrict__ stats)
{
  int c = blockIdx.x;  // 256 blocks, one per channel
  float s = 0.f, q = 0.f;
  for (int n = 0; n < 4; ++n) {
    const float* zp = z + ((size_t)n * CIN + c) * HW;
    for (int p = threadIdx.x; p < HW; p += 256) {
      float v = zp[p];
      s += v;
      q += v * v;
    }
  }
  __shared__ float ss[256], qq[256];
  ss[threadIdx.x] = s; qq[threadIdx.x] = q;
  __syncthreads();
  for (int st = 128; st > 0; st >>= 1) {
    if (threadIdx.x < st) {
      ss[threadIdx.x] += ss[threadIdx.x + st];
      qq[threadIdx.x] += qq[threadIdx.x + st];
    }
    __syncthreads();
  }
  if (threadIdx.x == 0) { stats[c] = ss[0]; stats[256 + c] = qq[0]; }
}

__global__ void bn_finalize_kernel(const float* __restrict__ stats,
                                   const float* __restrict__ gamma,
                                   const float* __restrict__ beta,
                                   float cnt_inv, float* __restrict__ affine)
{
  int c = threadIdx.x;  // 256 threads
  float mean = stats[c] * cnt_inv;
  float var  = stats[256 + c] * cnt_inv - mean * mean;
  float a    = gamma[c] * rsqrtf(var + 1e-5f);
  affine[c]       = a;
  affine[256 + c] = beta[c] - mean * a;
}

// ------------------------------------------------- 1x1 head conv (fused affine)

__global__ __launch_bounds__(256)
void head_kernel(const float* __restrict__ z, const float* __restrict__ affine,
                 const float* __restrict__ wh, const float* __restrict__ bh,
                 const float* __restrict__ scale_ptr, int scale_idx, int useExp,
                 int Cout, int HW, float* __restrict__ outBase, long long bStride)
{
  int idx = blockIdx.x * 256 + threadIdx.x;
  int total = 4 * Cout * HW;
  if (idx >= total) return;
  int p  = idx % HW;
  int co = (idx / HW) % Cout;
  int n  = idx / (HW * Cout);
  const float* zp = z + (size_t)n * CIN * HW + p;
  const float* wp = wh + (size_t)co * CIN;
  float acc = bh[co];
  for (int ci = 0; ci < CIN; ++ci)
    acc = fmaf(wp[ci], fmaf(affine[ci], zp[(size_t)ci * HW], affine[256 + ci]), acc);
  if (useExp) acc = expf(scale_ptr[scale_idx] * acc);
  outBase[(long long)n * bStride + (long long)co * HW + p] = acc;
}

// ----------------------------------------------------- boxes / scores / labels

__global__ __launch_bounds__(256)
void postprocess_kernel(float* __restrict__ d_out, int lvlOff, int HW, int shift,
                        int pixOff, float stride)
{
  int idx = blockIdx.x * 256 + threadIdx.x;
  if (idx >= 4 * HW) return;
  int p = idx & (HW - 1);
  int n = idx >> (2 * shift);
  const float* reg = d_out + (long long)n * 1827840LL + lvlOff;
  const float* lab = reg + 4 * HW;
  float l = reg[p], t = reg[HW + p], r = reg[2 * HW + p], b = reg[3 * HW + p];
  const int W = 1 << shift;
  float cx = ((p & (W - 1)) + 0.5f) * stride;
  float cy = ((p >> shift)  + 0.5f) * stride;
  long long gp = (long long)n * 21760 + pixOff + p;
  float* boxes = d_out + 7311360LL + gp * 4;
  boxes[0] = cx - l; boxes[1] = cy - t; boxes[2] = cx + r; boxes[3] = cy + b;
  float best = lab[p];
  int bi = 0;
  for (int c = 1; c < 80; ++c) {
    float v = lab[(size_t)c * HW + p];
    if (v > best) { best = v; bi = c; }
  }
  d_out[7659520LL + gp] = 1.f / (1.f + expf(-best));
  ((int*)d_out)[7746560LL + gp] = bi;
}

// =============================================================== host driver

extern "C" void kernel_launch(void* const* d_in, const int* in_sizes, int n_in,
                              void* d_out, int out_size, void* d_ws, size_t ws_size,
                              hipStream_t stream)
{
  (void)in_sizes; (void)n_in; (void)out_size; (void)ws_size;
  // d_in layout (setup_inputs leaf order):
  // 0..3 c1..c4 ; 4..11 cross (w,b)x4 ; 12..27 reg_branch (w,b,g,bt)x4 ;
  // 28..43 label_branch ; 44,45 reg_head ; 46,47 label_head ; 48 scale_params
  const float* c_in[4] = {(const float*)d_in[0], (const float*)d_in[1],
                          (const float*)d_in[2], (const float*)d_in[3]};

  char* base = (char*)d_ws;
  size_t off = 0;
  auto carve = [&](size_t bytes) -> void* {
    void* p = base + off;
    off += (bytes + 255) & ~(size_t)255;
    return p;
  };
  _Float16* w16[8];
  for (int j = 0; j < 8; ++j)
    w16[j] = (_Float16*)carve(sizeof(_Float16) * CIN * CIN * 9);
  int shifts[4] = {4, 5, 6, 7};  // levels d4,d3,d2,d1 : H=W=16,32,64,128
  float* dlv[4];
  for (int i = 0; i < 4; ++i)
    dlv[i] = (float*)carve(sizeof(float) * 4 * CIN * ((size_t)1 << (2 * shifts[i])));
  float* zb[2];
  zb[0] = (float*)carve(sizeof(float) * 4 * CIN * 16384);
  zb[1] = (float*)carve(sizeof(float) * 4 * CIN * 16384);
  float* stats = (float*)carve(sizeof(float) * 512);
  float* affId = (float*)carve(sizeof(float) * 512);
  float* aff[2] = {(float*)carve(sizeof(float) * 512), (float*)carve(sizeof(float) * 512)};

  // Convert+transpose 8 conv3x3 weight tensors to f16 [tap][co][ci].
  int nel = CIN * CIN * 9;
  for (int br = 0; br < 2; ++br)
    for (int k = 0; k < 4; ++k) {
      const float* w = (const float*)d_in[(br ? 28 : 12) + 4 * k];
      wconv_transpose_f16<<<(nel + 255) / 256, 256, 0, stream>>>(w, w16[br * 4 + k]);
    }
  init_affine_identity<<<1, 256, 0, stream>>>(affId);

  // FPN cross convs: level i=0 is d4 (input c4, cross_params[3]).
  int cinL[4] = {512, 256, 128, 64};
  for (int i = 0; i < 4; ++i) {
    int ci_idx = 3 - i;
    const float* w = (const float*)d_in[4 + 2 * ci_idx];
    const float* b = (const float*)d_in[5 + 2 * ci_idx];
    int sh = shifts[i];
    int total = 4 * CIN * (1 << (2 * sh));
    const float* up = (i == 0) ? nullptr : dlv[i - 1];
    cross_conv_kernel<<<(total + 255) / 256, 256, 0, stream>>>(
        c_in[ci_idx], w, b, up, dlv[i], cinL[i], sh, (i == 0) ? 0 : 1);
  }

  const float* scale_ptr = (const float*)d_in[48];
  long long lvlOff[4] = {0, 21504, 107520, 451584};
  int   pixOff[4]  = {0, 256, 1280, 5376};
  float strideL[4] = {32.f, 16.f, 8.f, 4.f};

  for (int i = 0; i < 4; ++i) {
    int sh = shifts[i];
    int HW = 1 << (2 * sh);
    for (int br = 0; br < 2; ++br) {
      int pb = br ? 28 : 12;
      const float* cur = dlv[i];
      for (int k = 0; k < 4; ++k) {
        const float* affIn = (k == 0) ? affId : aff[(k - 1) & 1];
        conv3x3_bnin_relu_wmma<<<dim3(CIN / MT, HW / NT, 4), 256, 0, stream>>>(
            cur, affIn, w16[br * 4 + k], (const float*)d_in[pb + 4 * k + 1],
            zb[k & 1], sh);
        bn_stats_kernel<<<256, 256, 0, stream>>>(zb[k & 1], HW, stats);
        bn_finalize_kernel<<<1, 256, 0, stream>>>(
            stats, (const float*)d_in[pb + 4 * k + 2],
            (const float*)d_in[pb + 4 * k + 3], 1.f / (float)(4 * HW), aff[k & 1]);
        cur = zb[k & 1];
      }
      int Cout = br ? 80 : 4;
      const float* wh = (const float*)d_in[br ? 46 : 44];
      const float* bh = (const float*)d_in[br ? 47 : 45];
      float* outBase = (float*)d_out + lvlOff[i] + (br ? 4 * HW : 0);
      int total = 4 * Cout * HW;
      head_kernel<<<(total + 255) / 256, 256, 0, stream>>>(
          zb[1], aff[1], wh, bh, scale_ptr, i, br ? 0 : 1, Cout, HW,
          outBase, 1827840LL);
    }
    int total = 4 * HW;
    postprocess_kernel<<<(total + 255) / 256, 256, 0, stream>>>(
        (float*)d_out, (int)lvlOff[i], HW, sh, pixOff[i], strideL[i]);
  }
}
